// QuantizedMambaModule_25537875542088
// MI455X (gfx1250) — matrix operand order repaired and checked by reading, not compiled, not run on hardware
//
#include <hip/hip_runtime.h>
#include <hip/hip_bf16.h>
#include <stdint.h>

// ---------------- constants ----------------
#define D_MODEL   768
#define D_STATE   16
#define D_CONV    4
#define D_INNER   1536   // 2 * D_MODEL
#define DT_RANK   48
#define DT_KPAD   64     // dt GEMM K padded to multiple of 32
#define BATCH     4
#define SEQLEN    4096
#define BL        (BATCH * SEQLEN)          // 16384 rows
#define XZ_COLS   (2 * D_INNER)             // 3072
#define XDBL_COLS (DT_RANK + 2 * D_STATE)   // 80
#define NCHUNK    64                        // scan chunks
#define CLEN      (SEQLEN / NCHUNK)         // 64 steps per chunk

typedef __bf16 bf16_t;
typedef __attribute__((ext_vector_type(16))) __bf16    v16bf;
typedef __attribute__((ext_vector_type(8)))  float     v8f;
typedef __attribute__((ext_vector_type(4)))  uint32_t  u32x4;

struct Frag128 { u32x4 lo, hi; };   // 32 bytes == v16bf

// fp32 -> bf16 (RNE) as raw bits
__device__ __forceinline__ uint32_t f2bf_bits(float f) {
    union { float f; uint32_t u; } cv; cv.f = f;
    uint32_t u = cv.u;
    return (u + 0x7FFFu + ((u >> 16) & 1u)) >> 16;
}
__device__ __forceinline__ bf16_t f2bf(float f) {
    uint16_t h = (uint16_t)f2bf_bits(f);
    return __builtin_bit_cast(bf16_t, h);
}
__device__ __forceinline__ bf16_t bf_zero() {
    return __builtin_bit_cast(bf16_t, (uint16_t)0);
}
__device__ __forceinline__ uint32_t pack2bf(float lo, float hi) {
    return f2bf_bits(lo) | (f2bf_bits(hi) << 16);
}
__device__ __forceinline__ float softplus_f(float x) {
    return (x > 20.f) ? x : log1pf(__expf(x));
}
__device__ __forceinline__ float silu_f(float x) {
    return x / (1.f + __expf(-x));
}

// P/F/H buffer index: (b, chunk, n, d) with d fastest for coalescing
__device__ __forceinline__ size_t pf_idx(int b, int c, int n, int d) {
    return (((size_t)(b * NCHUNK + c) * D_STATE + n) * D_INNER) + d;
}

// ======================================================================
// Packed fp32 -> bf16 conversion (8 elems/thread; sizes are multiples of 8)
// ======================================================================
__global__ __launch_bounds__(256) void f32_to_bf16_kernel(
    const float* __restrict__ in, bf16_t* __restrict__ out, long n)
{
    long i = ((long)blockIdx.x * blockDim.x + threadIdx.x) * 8;
    if (i >= n) return;
    float4 f0 = *(const float4*)(in + i);
    float4 f1 = *(const float4*)(in + i + 4);
    u32x4 o;
    o.x = pack2bf(f0.x, f0.y);
    o.y = pack2bf(f0.z, f0.w);
    o.z = pack2bf(f1.x, f1.y);
    o.w = pack2bf(f1.z, f1.w);
    *(u32x4*)(out + i) = o;
}

// dt_proj_w (1536 x 48 fp32) -> (1536 x 64 bf16), zero-padded cols 48..63
__global__ __launch_bounds__(256) void dtw_pad_kernel(
    const float* __restrict__ in, bf16_t* __restrict__ out)
{
    int i = blockIdx.x * blockDim.x + threadIdx.x;
    if (i >= D_INNER * DT_KPAD) return;
    const int r = i >> 6, c = i & (DT_KPAD - 1);
    out[i] = (c < DT_RANK) ? f2bf(in[r * DT_RANK + c]) : bf_zero();
}

// ======================================================================
// WMMA GEMM, bf16 inputs:  C[M,N] = A[M,K] * W[N,K]^T
// One wave computes a 16x64 tile (4 N-subtiles share the A fragment).
// All strides / trip counts compile-time; K % 32 == 0 (no tail path).
// MODE 0: fp32 store.  MODE 1: softplus(acc+bias[n]).
// MODE 2: fp32 store + bf16 dual-store of cols 0..63 into C2 (ld 64,
//         cols 48..63 written as zero) -- feeds the padded dt GEMM.
// ======================================================================
#define NSUB 4
template<int N, int K, int LDA, int LDW, int LDC, int MODE>
__global__ __launch_bounds__(256) void wmma_gemm_bf16(
    const bf16_t* __restrict__ A, const bf16_t* __restrict__ W,
    float* __restrict__ C, const float* __restrict__ bias,
    bf16_t* __restrict__ C2)
{
    constexpr int NTILES = (N + 15) / 16;
    constexpr int KTILES = K / 32;
    static_assert(K % 32 == 0, "K must be a multiple of 32");

    const int lane   = threadIdx.x & 31;
    const int wave   = threadIdx.x >> 5;
    const int mtile  = blockIdx.y * 8 + wave;   // grid.y = M/128
    const int ntile0 = blockIdx.x * NSUB;

    const int half = lane >> 4;                 // 0/1
    const int l16  = lane & 15;
    const int mrow = mtile * 16 + l16;

    // A fragment stream: lane holds K = {g*16 + half*8 + 0..7}, g = 0,1
    const bf16_t* __restrict__ ap = A + (size_t)mrow * LDA + half * 8;

    // B fragment streams: lane holds K = {g*16 + half*8 + 0..7}, g = 0,1
    const bf16_t* bp[NSUB];
    bool valid[NSUB];
    #pragma unroll
    for (int s = 0; s < NSUB; ++s) {
        const int nt = ntile0 + s;
        valid[s] = (nt < NTILES);
        const int nc = (valid[s] ? nt * 16 : 0) + l16;   // clamp to safe row
        bp[s] = W + (size_t)nc * LDW + half * 8;
    }

    v8f acc[NSUB] = {};
    #pragma unroll 2
    for (int kt = 0; kt < KTILES; ++kt) {
        const int kb = kt * 32;
        Frag128 fa;
        fa.lo = *(const u32x4*)(ap + kb);
        fa.hi = *(const u32x4*)(ap + kb + 16);
        const v16bf a = __builtin_bit_cast(v16bf, fa);
        #pragma unroll
        for (int s = 0; s < NSUB; ++s) {
            Frag128 fb;
            fb.lo = *(const u32x4*)(bp[s] + kb);
            fb.hi = *(const u32x4*)(bp[s] + kb + 16);
            const v16bf b = __builtin_bit_cast(v16bf, fb);
            acc[s] = __builtin_amdgcn_wmma_f32_16x16x32_bf16(
                         false, a, false, b, (short)0, acc[s], false, false);
        }
        // software prefetch two K-tiles ahead (global_prefetch_b8;
        // speculative: OOB translations are silently dropped)
        __builtin_prefetch(ap + kb + 64, 0, 0);
        __builtin_prefetch(bp[0] + kb + 64, 0, 0);
        __builtin_prefetch(bp[1] + kb + 64, 0, 0);
        __builtin_prefetch(bp[2] + kb + 64, 0, 0);
        __builtin_prefetch(bp[3] + kb + 64, 0, 0);
    }

    // C/D layout: VGPR r -> M = r + 8*half, N = lane%16
    #pragma unroll
    for (int s = 0; s < NSUB; ++s) {
        if (!valid[s]) continue;
        const int nout = (ntile0 + s) * 16 + l16;
        #pragma unroll
        for (int r = 0; r < 8; ++r) {
            const int mout = mtile * 16 + r + 8 * half;
            float v = acc[s][r];
            if constexpr (MODE == 1) v = softplus_f(v + bias[nout]);
            C[(size_t)mout * LDC + nout] = v;
            if constexpr (MODE == 2) {
                if (nout < DT_RANK)
                    C2[(size_t)mout * DT_KPAD + nout] = f2bf(v);
                else if (nout < DT_KPAD)
                    C2[(size_t)mout * DT_KPAD + nout] = bf_zero();
            }
        }
    }
}

// ======================================================================
// Depthwise causal conv1d (k=4, left pad 3) + bias + SiLU.
// Reads x half of xz; writes xc fp32 (scan input) and xc bf16 (GEMM A).
// ======================================================================
__global__ __launch_bounds__(256) void conv_silu_kernel(
    const float* __restrict__ xz, const float* __restrict__ conv_w,
    const float* __restrict__ conv_b, float* __restrict__ xc,
    bf16_t* __restrict__ xcb)
{
    size_t idx = (size_t)blockIdx.x * blockDim.x + threadIdx.x;
    if (idx >= (size_t)BL * D_INNER) return;
    const int d = (int)(idx % D_INNER);
    const size_t bl = idx / D_INNER;
    const int l = (int)(bl % SEQLEN);
    const size_t brow = bl - l;

    const float w0 = conv_w[d * D_CONV + 0];
    const float w1 = conv_w[d * D_CONV + 1];
    const float w2 = conv_w[d * D_CONV + 2];
    const float w3 = conv_w[d * D_CONV + 3];
    float s = conv_b[d];
    if (l >= 3) s += xz[(brow + l - 3) * XZ_COLS + d] * w0;
    if (l >= 2) s += xz[(brow + l - 2) * XZ_COLS + d] * w1;
    if (l >= 1) s += xz[(brow + l - 1) * XZ_COLS + d] * w2;
    s += xz[(brow + l) * XZ_COLS + d] * w3;
    const float r = silu_f(s);
    xc[idx]  = r;
    xcb[idx] = f2bf(r);
}

// ======================================================================
// Chunked selective scan (h_t = a_t h_{t-1} + w_t is linear ->
// over a chunk:  h_end = (prod a) * h_0 + F).  3 passes, 64 chunks.
// ======================================================================

// Pass 1: per (b, d, chunk): local scan from h=0; store per-state
// chunk product P[16] and chunk suffix F[16].
__global__ __launch_bounds__(256) void scan_pass1_kernel(
    const float* __restrict__ xc,     // (B,L,1536)  u
    const float* __restrict__ xdbl,   // (B,L,80)    [dt_low | B | C]
    const float* __restrict__ dt_in,  // (B,L,1536)
    const float* __restrict__ A_log,  // (1536,16)
    float* __restrict__ P, float* __restrict__ F)
{
    const int d = blockIdx.x * blockDim.x + threadIdx.x;
    const int c = blockIdx.y;
    const int b = blockIdx.z;
    if (d >= D_INNER) return;

    float Acoef[D_STATE];
    #pragma unroll
    for (int n = 0; n < D_STATE; ++n) Acoef[n] = -__expf(A_log[d * D_STATE + n]);

    float h[D_STATE], pr[D_STATE];
    #pragma unroll
    for (int n = 0; n < D_STATE; ++n) { h[n] = 0.f; pr[n] = 1.f; }

    const int l0 = c * CLEN;
    for (int l = l0; l < l0 + CLEN; ++l) {
        const size_t bl = (size_t)b * SEQLEN + l;
        const float dt = dt_in[bl * D_INNER + d];
        const float u  = xc[bl * D_INNER + d];
        const float* bc = xdbl + bl * XDBL_COLS + DT_RANK;
        const float du = dt * u;
        #pragma unroll
        for (int n = 0; n < D_STATE; ++n) {
            const float dA = __expf(dt * Acoef[n]);
            h[n]  = dA * h[n] + du * bc[n];
            pr[n] = dA * pr[n];
        }
    }
    #pragma unroll
    for (int n = 0; n < D_STATE; ++n) {
        P[pf_idx(b, c, n, d)] = pr[n];
        F[pf_idx(b, c, n, d)] = h[n];
    }
}

// Pass 2: per (b, n, d): serial combine over chunks;
// overwrite P[c] with H_c = incoming state of chunk c.
__global__ __launch_bounds__(256) void scan_pass2_kernel(
    float* __restrict__ P, const float* __restrict__ F)
{
    int i = blockIdx.x * blockDim.x + threadIdx.x;   // over B*D_STATE*D_INNER
    if (i >= BATCH * D_STATE * D_INNER) return;
    const int d = i % D_INNER;
    const int n = (i / D_INNER) % D_STATE;
    const int b = i / (D_INNER * D_STATE);

    float cur = 0.f;
    for (int c = 0; c < NCHUNK; ++c) {
        const size_t idx = pf_idx(b, c, n, d);
        const float pc = P[idx];
        const float fc = F[idx];
        P[idx] = cur;                 // H_c
        cur = fc + pc * cur;
    }
}

// Pass 3: per (b, d, chunk): re-run local scan seeded with H_c,
// produce y_t, add D*u skip, gate with silu(z), store bf16.
__global__ __launch_bounds__(256) void scan_pass3_kernel(
    const float* __restrict__ xc,     // (B,L,1536)  u
    const float* __restrict__ xdbl,   // (B,L,80)
    const float* __restrict__ dt_in,  // (B,L,1536)
    const float* __restrict__ xz,     // (B,L,3072)  z = cols 1536..
    const float* __restrict__ A_log,  // (1536,16)
    const float* __restrict__ Dvec,   // (1536)
    const float* __restrict__ H,      // chunk-incoming states (== P buffer)
    bf16_t*      __restrict__ y_bf)   // (B,L,1536)
{
    const int d = blockIdx.x * blockDim.x + threadIdx.x;
    const int c = blockIdx.y;
    const int b = blockIdx.z;
    if (d >= D_INNER) return;

    float Acoef[D_STATE];
    #pragma unroll
    for (int n = 0; n < D_STATE; ++n) Acoef[n] = -__expf(A_log[d * D_STATE + n]);
    const float Dd = Dvec[d];

    float h[D_STATE];
    #pragma unroll
    for (int n = 0; n < D_STATE; ++n) h[n] = H[pf_idx(b, c, n, d)];

    const int l0 = c * CLEN;
    for (int l = l0; l < l0 + CLEN; ++l) {
        const size_t bl = (size_t)b * SEQLEN + l;
        const float dt = dt_in[bl * D_INNER + d];
        const float u  = xc[bl * D_INNER + d];
        const float z  = xz[bl * XZ_COLS + D_INNER + d];
        const float* bc = xdbl + bl * XDBL_COLS + DT_RANK;
        const float du = dt * u;
        float y = 0.f;
        #pragma unroll
        for (int n = 0; n < D_STATE; ++n) {
            const float dA = __expf(dt * Acoef[n]);
            h[n] = dA * h[n] + du * bc[n];
            y += h[n] * bc[D_STATE + n];
        }
        y += Dd * u;
        y_bf[bl * D_INNER + d] = f2bf(y * silu_f(z));
    }
}

// ======================================================================
extern "C" void kernel_launch(void* const* d_in, const int* in_sizes, int n_in,
                              void* d_out, int out_size, void* d_ws, size_t ws_size,
                              hipStream_t stream) {
    (void)in_sizes; (void)n_in; (void)out_size; (void)ws_size;
    const float* hidden    = (const float*)d_in[0];  // (4,4096,768)
    const float* in_proj_w = (const float*)d_in[1];  // (3072,768)
    const float* conv_w    = (const float*)d_in[2];  // (1536,1,4)
    const float* conv_b    = (const float*)d_in[3];  // (1536)
    const float* x_proj_w  = (const float*)d_in[4];  // (80,1536)
    const float* dt_proj_w = (const float*)d_in[5];  // (1536,48)
    const float* dt_proj_b = (const float*)d_in[6];  // (1536)
    const float* A_log     = (const float*)d_in[7];  // (1536,16)
    const float* Dvec      = (const float*)d_in[8];  // (1536)
    const float* out_proj_w= (const float*)d_in[9];  // (768,1536)
    float* out = (float*)d_out;                      // (4,4096,768) fp32

    // ---------------- workspace carving (all regions 16B aligned) -------
    char* p = (char*)d_ws;
    float* xz    = (float*)p;  p += (size_t)BL * XZ_COLS   * 4;   // 201.3 MB
    float* xc    = (float*)p;  p += (size_t)BL * D_INNER   * 4;   // 100.7 MB
    float* dt    = (float*)p;  p += (size_t)BL * D_INNER   * 4;   // 100.7 MB
    float* xdbl  = (float*)p;  p += (size_t)BL * XDBL_COLS * 4;   //   5.2 MB
    float* Pbuf  = (float*)p;  p += (size_t)BATCH * NCHUNK * D_STATE * D_INNER * 4; // 25.2 MB
    float* Fbuf  = (float*)p;  p += (size_t)BATCH * NCHUNK * D_STATE * D_INNER * 4; // 25.2 MB
    bf16_t* hid_bf   = (bf16_t*)p; p += (size_t)BL * D_MODEL   * 2;
    bf16_t* xc_bf    = (bf16_t*)p; p += (size_t)BL * D_INNER   * 2;
    bf16_t* dtlow_bf = (bf16_t*)p; p += (size_t)BL * DT_KPAD   * 2;  // padded dt_low
    bf16_t* y_bf     = (bf16_t*)p; p += (size_t)BL * D_INNER   * 2;
    bf16_t* w_in_bf  = (bf16_t*)p; p += (size_t)XZ_COLS   * D_MODEL * 2;
    bf16_t* w_x_bf   = (bf16_t*)p; p += (size_t)XDBL_COLS * D_INNER * 2;
    bf16_t* w_dt_bf  = (bf16_t*)p; p += (size_t)D_INNER   * DT_KPAD * 2; // padded
    bf16_t* w_out_bf = (bf16_t*)p; p += (size_t)D_MODEL   * D_INNER * 2;

    dim3 blk(256, 1, 1);
    auto cvt = [&](const float* src, bf16_t* dst, long n) {
        dim3 g((unsigned)((n / 8 + 255) / 256), 1, 1);
        f32_to_bf16_kernel<<<g, blk, 0, stream>>>(src, dst, n);
    };

    // 0) one-time bf16 staging of hidden + weights (dt weight zero-padded)
    cvt(hidden,     hid_bf,   (long)BL * D_MODEL);
    cvt(in_proj_w,  w_in_bf,  (long)XZ_COLS * D_MODEL);
    cvt(x_proj_w,   w_x_bf,   (long)XDBL_COLS * D_INNER);
    cvt(out_proj_w, w_out_bf, (long)D_MODEL * D_INNER);
    {
        dim3 g((D_INNER * DT_KPAD + 255) / 256, 1, 1);
        dtw_pad_kernel<<<g, blk, 0, stream>>>(dt_proj_w, w_dt_bf);
    }

    const unsigned gy = BL / 128;   // 128 blocks of 8 waves -> 1024 M-tiles

    // 1) in_proj: xz = hidden * in_proj_w^T   (16384x3072, K=768)
    {
        dim3 grid((XZ_COLS / 16) / NSUB, gy, 1);
        wmma_gemm_bf16<XZ_COLS, D_MODEL, D_MODEL, D_MODEL, XZ_COLS, 0>
            <<<grid, blk, 0, stream>>>(hid_bf, w_in_bf, xz, nullptr, nullptr);
    }
    // 2) causal depthwise conv + SiLU -> xc (fp32 + bf16)
    {
        size_t tot = (size_t)BL * D_INNER;
        dim3 grid((unsigned)((tot + 255) / 256), 1, 1);
        conv_silu_kernel<<<grid, blk, 0, stream>>>(xz, conv_w, conv_b, xc, xc_bf);
    }
    // 3) x_proj: xdbl = xc * x_proj_w^T (16384x80, K=1536); bf16 dt_low dual-store
    {
        dim3 grid((XDBL_COLS / 16 + NSUB - 1) / NSUB, gy, 1);
        wmma_gemm_bf16<XDBL_COLS, D_INNER, D_INNER, D_INNER, XDBL_COLS, 2>
            <<<grid, blk, 0, stream>>>(xc_bf, w_x_bf, xdbl, nullptr, dtlow_bf);
    }
    // 4) dt = softplus(dt_low * dt_proj_w^T + b)  (16384x1536, K=64 padded)
    {
        dim3 grid((D_INNER / 16) / NSUB, gy, 1);
        wmma_gemm_bf16<D_INNER, DT_KPAD, DT_KPAD, DT_KPAD, D_INNER, 1>
            <<<grid, blk, 0, stream>>>(dtlow_bf, w_dt_bf, dt, dt_proj_b, nullptr);
    }
    // 5) chunked selective scan (3 passes) -> y_bf
    {
        dim3 grid1(D_INNER / 256, NCHUNK, BATCH);
        scan_pass1_kernel<<<grid1, blk, 0, stream>>>(xc, xdbl, dt, A_log, Pbuf, Fbuf);
        dim3 grid2((BATCH * D_STATE * D_INNER) / 256, 1, 1);
        scan_pass2_kernel<<<grid2, blk, 0, stream>>>(Pbuf, Fbuf);
        dim3 grid3(D_INNER / 256, NCHUNK, BATCH);
        scan_pass3_kernel<<<grid3, blk, 0, stream>>>(xc, xdbl, dt, xz, A_log, Dvec,
                                                     Pbuf, y_bf);
    }
    // 6) out_proj: out = y * out_proj_w^T     (16384x768, K=1536)
    {
        dim3 grid((D_MODEL / 16) / NSUB, gy, 1);
        wmma_gemm_bf16<D_MODEL, D_INNER, D_INNER, D_INNER, D_MODEL, 0>
            <<<grid, blk, 0, stream>>>(y_bf, w_out_bf, out, nullptr, nullptr);
    }
}
